// MultiHeadAttention_26654567039072
// MI455X (gfx1250) — matrix operand order repaired and checked by reading, compiled, not run on hardware
//
#include <hip/hip_runtime.h>
#include <hip/hip_bf16.h>

typedef _Float16 v16h __attribute__((ext_vector_type(16)));
typedef _Float16 v8h  __attribute__((ext_vector_type(8)));
typedef _Float16 v2h  __attribute__((ext_vector_type(2)));
typedef float    v8f  __attribute__((ext_vector_type(8)));
typedef float    v4f  __attribute__((ext_vector_type(4)));

#define D_MODEL 1024
#define S_LEN   2048
#define NH      16
#define DK      64
#define BATCH   2

__device__ __forceinline__ v8f wmma_f32_f16(v16h a, v16h b, v8f c) {
  return __builtin_amdgcn_wmma_f32_16x16x32_f16(false, a, false, b, (short)0, c, false, false);
}

// A-fragment (16x32 f16, M x K): lane holds row M=lane%16, K = hs*8+{0..7} and 16+hs*8+{0..7}
__device__ __forceinline__ v16h lds_afrag(const _Float16* rowp, int hs) {
  v8h x = *(const v8h*)(rowp + hs * 8);
  v8h y = *(const v8h*)(rowp + 16 + hs * 8);
  v16h r;
#pragma unroll
  for (int i = 0; i < 8; ++i) { r[i] = x[i]; r[i + 8] = y[i]; }
  return r;
}

// B-fragment (32x16 f16, K x N): lane holds col N=lane%16, K = hs*8+{0..7} and 16+hs*8+{0..7}
__device__ __forceinline__ v16h lds_bfrag(const _Float16* rowp, int hs) {
  v8h x = *(const v8h*)(rowp + hs * 8);
  v8h y = *(const v8h*)(rowp + 16 + hs * 8);
  v16h r;
#pragma unroll
  for (int i = 0; i < 8; ++i) { r[i] = x[i]; r[i + 8] = y[i]; }
  return r;
}

// convert 16 f32 (in regs) -> 16 f16, store as two 16B LDS writes
__device__ __forceinline__ void cvt16_store_regs(v4f f0, v4f f1, v4f f2, v4f f3,
                                                 _Float16* dst) {
  v8h h0, h1;
#pragma unroll
  for (int i = 0; i < 4; ++i) {
    h0[i]     = (_Float16)f0[i];
    h0[i + 4] = (_Float16)f1[i];
    h1[i]     = (_Float16)f2[i];
    h1[i + 4] = (_Float16)f3[i];
  }
  ((v8h*)dst)[0] = h0;
  ((v8h*)dst)[1] = h1;
}

__device__ __forceinline__ void cvt16_store(const float* __restrict__ g, _Float16* dst) {
  cvt16_store_regs(((const v4f*)g)[0], ((const v4f*)g)[1],
                   ((const v4f*)g)[2], ((const v4f*)g)[3], dst);
}

// Async copy of 64B (global -> LDS), no VGPR data round-trip; tracked by ASYNCcnt.
// lds byte address = low 32 bits of the generic shared-aperture pointer.
__device__ __forceinline__ void async_copy_64B(const _Float16* g, _Float16* lds) {
  unsigned loff = (unsigned)(unsigned long long)lds;
  asm volatile(
      "global_load_async_to_lds_b128 %0, %1, off\n\t"
      "global_load_async_to_lds_b128 %0, %1, off offset:16\n\t"
      "global_load_async_to_lds_b128 %0, %1, off offset:32\n\t"
      "global_load_async_to_lds_b128 %0, %1, off offset:48"
      :: "v"(loff), "v"(g) : "memory");
}

__device__ __forceinline__ void wait_async0() {
  asm volatile("s_wait_asynccnt 0x0" ::: "memory");
}

// ---------------------------------------------------------------------------
// C[M,N] = A[M,K] @ W[N,K]^T + bias ; A fp32-or-f16, W fp32, C f16-or-f32.
// Block tile 128x256x32, 256 threads = 8 waves (2x4), wave tile 64x64
// (16 WMMAs per wave per k-step). Double-buffered LDS, one barrier/step,
// global loads of the next k-slab issued before the WMMA block.
// ---------------------------------------------------------------------------
constexpr int GBM = 128, GBN = 256, GBK = 32, GP = GBK + 8;  // pitch 40 halves = 80B

template <bool A_HALF, bool OUT_HALF>
__global__ __launch_bounds__(256) void gemm_xwT(const void* __restrict__ Ain,
                                                const float* __restrict__ W,
                                                const float* __restrict__ bias,
                                                void* __restrict__ Cout,
                                                int M, int N, int K) {
  __shared__ _Float16 As[2][GBM * GP];   // 2 x 10KB
  __shared__ _Float16 Bs[2][GBN * GP];   // 2 x 20KB

  const int tid  = threadIdx.x;
  const int lane = tid & 31;
  const int wave = tid >> 5;
  const int wm   = wave & 1;        // 0..1 -> 64 rows each
  const int wn   = wave >> 1;       // 0..3 -> 64 cols each
  const int hs   = lane >> 4;
  const int lr   = lane & 15;
  const int bm0  = blockIdx.y * GBM;
  const int bn0  = blockIdx.x * GBN;

  const int lrow = tid >> 1;        // 0..127
  const int lcol = (tid & 1) * 16;  // 0 or 16

  v8f acc[4][4];
#pragma unroll
  for (int i = 0; i < 4; ++i)
#pragma unroll
    for (int j = 0; j < 4; ++j) acc[i][j] = (v8f)(0.f);

  // ---- stage k-slab 0 into buffer 0 ----
  if (A_HALF) {
    const _Float16* g = (const _Float16*)Ain + (size_t)(bm0 + lrow) * K + lcol;
    ((v8h*)&As[0][lrow * GP + lcol])[0] = ((const v8h*)g)[0];
    ((v8h*)&As[0][lrow * GP + lcol])[1] = ((const v8h*)g)[1];
  } else {
    cvt16_store((const float*)Ain + (size_t)(bm0 + lrow) * K + lcol,
                &As[0][lrow * GP + lcol]);
  }
#pragma unroll
  for (int j = 0; j < 2; ++j) {
    const int r = j * 128 + lrow;
    cvt16_store(W + (size_t)(bn0 + r) * K + lcol, &Bs[0][r * GP + lcol]);
  }
  __syncthreads();

  int cur = 0;
  for (int k0 = 0; k0 < K; k0 += GBK) {
    const bool more = (k0 + GBK) < K;
    const int  nxt  = cur ^ 1;

    // ---- issue global loads for the next k-slab (latency hidden by WMMAs) ----
    v4f raf[4];
    v8h rah[2];
    v4f rb[2][4];
    if (more) {
      const int kn = k0 + GBK;
      if (A_HALF) {
        const _Float16* g = (const _Float16*)Ain + (size_t)(bm0 + lrow) * K + kn + lcol;
        rah[0] = ((const v8h*)g)[0];
        rah[1] = ((const v8h*)g)[1];
      } else {
        const float* g = (const float*)Ain + (size_t)(bm0 + lrow) * K + kn + lcol;
#pragma unroll
        for (int i = 0; i < 4; ++i) raf[i] = ((const v4f*)g)[i];
      }
#pragma unroll
      for (int j = 0; j < 2; ++j) {
        const float* g = W + (size_t)(bn0 + j * 128 + lrow) * K + kn + lcol;
#pragma unroll
        for (int i = 0; i < 4; ++i) rb[j][i] = ((const v4f*)g)[i];
      }
    }

    // ---- compute on current buffer: 16 WMMAs per wave ----
    v16h af[4], bf[4];
#pragma unroll
    for (int mf = 0; mf < 4; ++mf)
      af[mf] = lds_afrag(&As[cur][(wm * 64 + mf * 16 + lr) * GP], hs);
#pragma unroll
    for (int nf = 0; nf < 4; ++nf)
      bf[nf] = lds_bfrag(&Bs[cur][(wn * 64 + nf * 16 + lr) * GP], hs);
#pragma unroll
    for (int mf = 0; mf < 4; ++mf)
#pragma unroll
      for (int nf = 0; nf < 4; ++nf)
        acc[mf][nf] = wmma_f32_f16(af[mf], bf[nf], acc[mf][nf]);

    // ---- commit prefetched slab into the alternate buffer ----
    if (more) {
      if (A_HALF) {
        ((v8h*)&As[nxt][lrow * GP + lcol])[0] = rah[0];
        ((v8h*)&As[nxt][lrow * GP + lcol])[1] = rah[1];
      } else {
        cvt16_store_regs(raf[0], raf[1], raf[2], raf[3], &As[nxt][lrow * GP + lcol]);
      }
#pragma unroll
      for (int j = 0; j < 2; ++j)
        cvt16_store_regs(rb[j][0], rb[j][1], rb[j][2], rb[j][3],
                         &Bs[nxt][(j * 128 + lrow) * GP + lcol]);
    }
    __syncthreads();
    cur = nxt;
  }

  // ---- epilogue: bias add, store ----
#pragma unroll
  for (int nf = 0; nf < 4; ++nf) {
    const int col = bn0 + wn * 64 + nf * 16 + lr;
    const float bv = bias[col];
#pragma unroll
    for (int mf = 0; mf < 4; ++mf) {
#pragma unroll
      for (int r = 0; r < 8; ++r) {
        const int row = bm0 + wm * 64 + mf * 16 + r + hs * 8;
        const float vv = acc[mf][nf][r] + bv;
        if (OUT_HALF)
          ((_Float16*)Cout)[(size_t)row * N + col] = (_Float16)vv;
        else
          ((float*)Cout)[(size_t)row * N + col] = vv;
      }
    }
  }
}

// ---------------------------------------------------------------------------
// Flash attention, causal. One WG per (b, h, 64-query tile); 4 waves;
// each wave owns 16 query rows. q/k/v are f16 [B*S, D_MODEL], head h at
// column offset h*DK. Q/K tiles staged with GLOBAL_LOAD_ASYNC_TO_LDS_B128.
// ---------------------------------------------------------------------------
__global__ __launch_bounds__(128) void flash_attn(const _Float16* __restrict__ q,
                                                  const _Float16* __restrict__ k,
                                                  const _Float16* __restrict__ v,
                                                  _Float16* __restrict__ o) {
  constexpr int QT = 64, KT = 64, P = DK + 8;  // pitch 72 halves = 144B
  __shared__ _Float16 Qs[QT * P];
  __shared__ _Float16 Ks[KT * P];
  __shared__ _Float16 Vt[DK * P];      // transposed: Vt[d][kk]
  __shared__ _Float16 Ps[4 * 16 * P];  // per-wave P staging

  const int tid  = threadIdx.x;
  const int lane = tid & 31;
  const int wv   = tid >> 5;
  const int hs   = lane >> 4;
  const int lr   = lane & 15;

  const int nqt = S_LEN / QT;  // 32
  const int qt  = blockIdx.x % nqt;
  const int h   = (blockIdx.x / nqt) % NH;
  const int b   = blockIdx.x / (nqt * NH);

  const size_t base = (size_t)b * S_LEN * D_MODEL + (size_t)h * DK;
  const _Float16* qg = q + base;
  const _Float16* kg = k + base;
  const _Float16* vg = v + base;

  const int srow = tid >> 1;           // staging: 2 threads per row
  const int scol = (tid & 1) * 32;     // 32 halves = 64B per thread

  // ---- async-stage Q tile: 64 rows x 64 halves ----
  async_copy_64B(qg + (size_t)(qt * QT + srow) * D_MODEL + scol, &Qs[srow * P + scol]);
  wait_async0();
  __syncthreads();

  v16h aQ[2];
  aQ[0] = lds_afrag(&Qs[(wv * 16 + lr) * P], hs);       // K 0..31
  aQ[1] = lds_afrag(&Qs[(wv * 16 + lr) * P + 32], hs);  // K 32..63

  v8f out[4];
#pragma unroll
  for (int i = 0; i < 4; ++i) out[i] = (v8f)(0.f);
  float mrun[8], lrun[8];
#pragma unroll
  for (int r = 0; r < 8; ++r) { mrun[r] = -3.0e38f; lrun[r] = 0.f; }

  for (int kt = 0; kt <= qt; ++kt) {
    __syncthreads();  // previous tile's LDS reads are done
    // ---- async-stage K tile ----
    async_copy_64B(kg + (size_t)(kt * KT + srow) * D_MODEL + scol, &Ks[srow * P + scol]);
    // ---- load V tile, store transposed with packed 2xf16 (b32) LDS writes ----
    {
      const int row0 = (tid >> 2) * 2;      // even row of a row pair
      const int d0   = (tid & 3) * 16;      // 16 d-columns per thread
      const _Float16* g0 = vg + (size_t)(kt * KT + row0) * D_MODEL + d0;
      const _Float16* g1 = g0 + D_MODEL;
      v8h a0 = ((const v8h*)g0)[0], a1 = ((const v8h*)g0)[1];
      v8h b0 = ((const v8h*)g1)[0], b1 = ((const v8h*)g1)[1];
#pragma unroll
      for (int j = 0; j < 8; ++j) {
        v2h p0; p0[0] = a0[j]; p0[1] = b0[j];
        *(v2h*)&Vt[(d0 + j) * P + row0] = p0;
        v2h p1; p1[0] = a1[j]; p1[1] = b1[j];
        *(v2h*)&Vt[(d0 + 8 + j) * P + row0] = p1;
      }
    }
    // ---- prefetch next K/V tiles into cache while we crunch this one ----
    if (kt + 1 <= qt) {
      const size_t off = (size_t)((kt + 1) * KT + srow) * D_MODEL + scol;
      __builtin_prefetch(kg + off, 0, 3);
      __builtin_prefetch(vg + off, 0, 3);
    }
    wait_async0();
    __syncthreads();

    const bool diag = (kt == qt);
    v8f sc[4];
    float mt[8];
#pragma unroll
    for (int r = 0; r < 8; ++r) mt[r] = -3.0e38f;

    // ---- S = Q K^T * (1/sqrt(dk)), causal mask on diagonal tile ----
#pragma unroll
    for (int nf = 0; nf < 4; ++nf) {
      v16h bK0 = lds_bfrag(&Ks[(nf * 16 + lr) * P], hs);
      v16h bK1 = lds_bfrag(&Ks[(nf * 16 + lr) * P + 32], hs);
      v8f s = (v8f)(0.f);
      s = wmma_f32_f16(aQ[0], bK0, s);
      s = wmma_f32_f16(aQ[1], bK1, s);
#pragma unroll
      for (int r = 0; r < 8; ++r) {
        float x = s[r] * 0.125f;  // 1/sqrt(64)
        if (diag) {
          const int kc = kt * KT + nf * 16 + lr;
          const int qr = qt * QT + wv * 16 + r + hs * 8;
          if (kc > qr) x = -1.0e30f;
        }
        sc[nf][r] = x;
        mt[r] = fmaxf(mt[r], x);
      }
    }
    // ---- row max across the 16 N-lanes ----
#pragma unroll
    for (int r = 0; r < 8; ++r) {
      mt[r] = fmaxf(mt[r], __shfl_xor(mt[r], 1, 32));
      mt[r] = fmaxf(mt[r], __shfl_xor(mt[r], 2, 32));
      mt[r] = fmaxf(mt[r], __shfl_xor(mt[r], 4, 32));
      mt[r] = fmaxf(mt[r], __shfl_xor(mt[r], 8, 32));
    }
    float corr[8], rs[8];
#pragma unroll
    for (int r = 0; r < 8; ++r) {
      const float mn = fmaxf(mrun[r], mt[r]);
      corr[r] = __expf(mrun[r] - mn);
      mrun[r] = mn;
      rs[r] = 0.f;
    }
    // ---- P = exp(S - m), stage to LDS row-major for A-operand reload ----
#pragma unroll
    for (int nf = 0; nf < 4; ++nf) {
#pragma unroll
      for (int r = 0; r < 8; ++r) {
        const float p = __expf(sc[nf][r] - mrun[r]);
        rs[r] += p;
        Ps[(wv * 16 + r + hs * 8) * P + nf * 16 + lr] = (_Float16)p;
      }
    }
#pragma unroll
    for (int r = 0; r < 8; ++r) {
      rs[r] += __shfl_xor(rs[r], 1, 32);
      rs[r] += __shfl_xor(rs[r], 2, 32);
      rs[r] += __shfl_xor(rs[r], 4, 32);
      rs[r] += __shfl_xor(rs[r], 8, 32);
      lrun[r] = lrun[r] * corr[r] + rs[r];
    }
#pragma unroll
    for (int nf = 0; nf < 4; ++nf)
#pragma unroll
      for (int r = 0; r < 8; ++r) out[nf][r] *= corr[r];

    asm volatile("s_wait_dscnt 0" ::: "memory");  // our P stores -> our reads
    v16h aP0 = lds_afrag(&Ps[(wv * 16 + lr) * P], hs);
    v16h aP1 = lds_afrag(&Ps[(wv * 16 + lr) * P + 32], hs);
#pragma unroll
    for (int nf = 0; nf < 4; ++nf) {
      v16h bV0 = lds_bfrag(&Vt[(nf * 16 + lr) * P], hs);
      v16h bV1 = lds_bfrag(&Vt[(nf * 16 + lr) * P + 32], hs);
      out[nf] = wmma_f32_f16(aP0, bV0, out[nf]);
      out[nf] = wmma_f32_f16(aP1, bV1, out[nf]);
    }
  }

  // ---- normalize and write ----
#pragma unroll
  for (int nf = 0; nf < 4; ++nf) {
#pragma unroll
    for (int r = 0; r < 8; ++r) {
      const int row = qt * QT + wv * 16 + r + hs * 8;
      const size_t idx = (size_t)(b * S_LEN + row) * D_MODEL + h * DK + nf * 16 + lr;
      o[idx] = (_Float16)(out[nf][r] / lrun[r]);
    }
  }
}

// ---------------------------------------------------------------------------
extern "C" void kernel_launch(void* const* d_in, const int* in_sizes, int n_in,
                              void* d_out, int out_size, void* d_ws, size_t ws_size,
                              hipStream_t stream) {
  (void)in_sizes; (void)n_in; (void)out_size; (void)ws_size;
  const float* Q  = (const float*)d_in[0];
  const float* K  = (const float*)d_in[1];
  const float* V  = (const float*)d_in[2];
  // d_in[3] = mask: causal, reconstructed analytically in flash_attn
  const float* Wq = (const float*)d_in[4];
  const float* bq = (const float*)d_in[5];
  const float* Wk = (const float*)d_in[6];
  const float* bk = (const float*)d_in[7];
  const float* Wv = (const float*)d_in[8];
  const float* bv = (const float*)d_in[9];
  const float* Wo = (const float*)d_in[10];
  const float* bo = (const float*)d_in[11];

  const int M = BATCH * S_LEN;  // 4096
  _Float16* q16 = (_Float16*)d_ws;
  _Float16* k16 = q16 + (size_t)M * D_MODEL;
  _Float16* v16 = k16 + (size_t)M * D_MODEL;
  _Float16* a16 = v16 + (size_t)M * D_MODEL;

  dim3 gg(D_MODEL / GBN, M / GBM);  // (4, 32)
  dim3 gb(256);
  gemm_xwT<false, true><<<gg, gb, 0, stream>>>(Q, Wq, bq, q16, M, D_MODEL, D_MODEL);
  gemm_xwT<false, true><<<gg, gb, 0, stream>>>(K, Wk, bk, k16, M, D_MODEL, D_MODEL);
  gemm_xwT<false, true><<<gg, gb, 0, stream>>>(V, Wv, bv, v16, M, D_MODEL, D_MODEL);

  flash_attn<<<BATCH * NH * (S_LEN / 64), 128, 0, stream>>>(q16, k16, v16, a16);

  gemm_xwT<true, false><<<gg, gb, 0, stream>>>(a16, Wo, bo, d_out, M, D_MODEL, D_MODEL);
}
